// RawEmbeddingKVAttention_1520418423116
// MI455X (gfx1250) — matrix operand order repaired and checked by reading, not compiled, not run on hardware
//
#include <hip/hip_runtime.h>
#include <hip/hip_bf16.h>

// ---------------------------------------------------------------------------
// Problem constants (B=1): S=4096 seq, D=512 model, H=8 heads, HD=64, top-K=8
// ---------------------------------------------------------------------------
constexpr int   cS     = 4096;
constexpr int   cD     = 512;
constexpr int   cH     = 8;
constexpr int   cHD    = 64;
constexpr int   cTOPK  = 8;
constexpr float cSCALE = 0.125f;            // 1/sqrt(64)

typedef __attribute__((ext_vector_type(16))) __bf16 v16bf;
typedef __attribute__((ext_vector_type(8)))  float  v8f;

union Frag16 { v16bf v; uint4 q[2]; };      // 32 bytes: one 16-bit WMMA operand fragment

__device__ __forceinline__ unsigned short f2bf(float f) {
    unsigned u = __float_as_uint(f);
    u += 0x7fffu + ((u >> 16) & 1u);        // round-to-nearest-even
    return (unsigned short)(u >> 16);
}
__device__ __forceinline__ float bf2f(unsigned short b) {
    return __uint_as_float((unsigned)b << 16);
}

// ---------------------------------------------------------------------------
// fp32 -> bf16 bulk convert
// ---------------------------------------------------------------------------
__global__ void cvt_f32_bf16(const float* __restrict__ in,
                             unsigned short* __restrict__ out, int n) {
    int i = blockIdx.x * blockDim.x + threadIdx.x;
    if (i < n) out[i] = f2bf(in[i]);
}

// ---------------------------------------------------------------------------
// y = x * W^T + b   (torch Linear layout: W is [N_out x K_in], row-major)
// A: [M x Kd] bf16, W: [N x Kd] bf16, out: f32 or bf16 [M x N]
// One 16x64 output strip per wave: the A fragment is loaded once per K-step
// and reused by 4 independent v_wmma_f32_16x16x32_bf16 (4 accumulators),
// quadrupling A reuse and giving the XDL pipe 4 dependence-free WMMAs per
// clause of b128 loads.
// ---------------------------------------------------------------------------
__global__ __launch_bounds__(256) void gemm_bf16_wmma(
    const unsigned short* __restrict__ A,
    const unsigned short* __restrict__ W,
    const float* __restrict__ bias,
    float* __restrict__ outF,
    unsigned short* __restrict__ outB,
    int M, int N, int Kd, int out_bf16)
{
    const int lane = threadIdx.x & 31;
    const int wave = threadIdx.x >> 5;                 // 8 waves / block
    const int tile = blockIdx.x * 8 + wave;            // 16x64 strips
    const int ntn  = N >> 6;                           // strips along N
    const int tm   = tile / ntn;
    const int tn   = tile - tm * ntn;
    if (tm >= (M >> 4)) return;                        // wave-uniform
    const int m0 = tm << 4, n0 = tn << 6;
    const int row = lane & 15;                         // A row / B column owned by lane
    const int lh  = lane >> 4;                         // lane half (ISA 16-bit layout)

    const unsigned short* arow = A + (size_t)(m0 + row) * Kd;
    const unsigned short* wr0  = W + (size_t)(n0 +  0 + row) * Kd;
    const unsigned short* wr1  = W + (size_t)(n0 + 16 + row) * Kd;
    const unsigned short* wr2  = W + (size_t)(n0 + 32 + row) * Kd;
    const unsigned short* wr3  = W + (size_t)(n0 + 48 + row) * Kd;

    v8f c0 = {0.f,0.f,0.f,0.f,0.f,0.f,0.f,0.f};
    v8f c1 = c0, c2 = c0, c3 = c0;
    for (int k0 = 0; k0 < Kd; k0 += 32) {
        Frag16 a, b0, b1, b2, b3;
        // A 16x32: lane half selects K {lh*8..+7} and {16+lh*8..+7}
        a.q[0] = *(const uint4*)(arow + k0 + lh * 8);
        a.q[1] = *(const uint4*)(arow + k0 + 16 + lh * 8);
        // B 32x16: column = W row; lane half selects K {lh*8..+7} and {16+lh*8..+7}
        b0.q[0] = *(const uint4*)(wr0 + k0 + lh * 8);
        b0.q[1] = *(const uint4*)(wr0 + k0 + 16 + lh * 8);
        b1.q[0] = *(const uint4*)(wr1 + k0 + lh * 8);
        b1.q[1] = *(const uint4*)(wr1 + k0 + 16 + lh * 8);
        b2.q[0] = *(const uint4*)(wr2 + k0 + lh * 8);
        b2.q[1] = *(const uint4*)(wr2 + k0 + 16 + lh * 8);
        b3.q[0] = *(const uint4*)(wr3 + k0 + lh * 8);
        b3.q[1] = *(const uint4*)(wr3 + k0 + 16 + lh * 8);
        if (k0 + 64 <= Kd) {                           // global_prefetch_b8 next slab
            __builtin_prefetch(arow + k0 + 64, 0, 3);
            __builtin_prefetch(wr0  + k0 + 64, 0, 3);
            __builtin_prefetch(wr2  + k0 + 64, 0, 3);
        }
        c0 = __builtin_amdgcn_wmma_f32_16x16x32_bf16(false, a.v, false, b0.v, (short)0, c0, false, false);
        c1 = __builtin_amdgcn_wmma_f32_16x16x32_bf16(false, a.v, false, b1.v, (short)0, c1, false, false);
        c2 = __builtin_amdgcn_wmma_f32_16x16x32_bf16(false, a.v, false, b2.v, (short)0, c2, false, false);
        c3 = __builtin_amdgcn_wmma_f32_16x16x32_bf16(false, a.v, false, b3.v, (short)0, c3, false, false);
    }
#pragma unroll
    for (int j = 0; j < 4; ++j) {
        const v8f& c = (j == 0) ? c0 : (j == 1) ? c1 : (j == 2) ? c2 : c3;
        const int ncol = n0 + j * 16 + (lane & 15);
        const float bb = bias ? bias[ncol] : 0.f;
#pragma unroll
        for (int r = 0; r < 8; ++r) {
            const int rr  = m0 + r + lh * 8;           // C/D layout: M = r + 8*half
            const float v = c[r] + bb;
            if (out_bf16) outB[(size_t)rr * N + ncol] = f2bf(v);
            else          outF[(size_t)rr * N + ncol] = v;
        }
    }
}

// ---------------------------------------------------------------------------
// Streaming causal top-8 attention. Grid: (S/16) blocks x 256 thr.
// Wave w = head w handles one 16-query tile: walks key tiles 0..qt,
// 2 WMMAs per 16x16 score tile (HD=64 = 2 x K32), stages scores through LDS.
// All 32 lanes scan: lanes L and L+16 co-own query row L%16 (8 columns each,
// private top-8, guarded bubble insert), merged once at the end via
// __shfl_xor(...,16). Then 8-entry softmax + V-row gather.
// ---------------------------------------------------------------------------
__global__ __launch_bounds__(256) void sparse_attn_wmma(
    const unsigned short* __restrict__ Qb,
    const unsigned short* __restrict__ Kb,
    const unsigned short* __restrict__ Vb,
    unsigned short* __restrict__ Ob)
{
    __shared__ float sc[cH][256];                      // 16x16 f32 tile per wave, 8 KB
    const int lane = threadIdx.x & 31;
    const int h    = threadIdx.x >> 5;                 // head == wave id
    const int qt   = blockIdx.x;
    const int q0   = qt << 4;
    const int row  = lane & 15;
    const int lh   = lane >> 4;

    // Q tile 16x64 -> two A fragments (K = 0..31, 32..63 of head dim)
    const unsigned short* qrow = Qb + (size_t)(q0 + row) * cD + h * cHD;
    Frag16 a0, a1;
    a0.q[0] = *(const uint4*)(qrow + lh * 8);
    a0.q[1] = *(const uint4*)(qrow + 16 + lh * 8);
    a1.q[0] = *(const uint4*)(qrow + 32 + lh * 8);
    a1.q[1] = *(const uint4*)(qrow + 48 + lh * 8);

    float ts[cTOPK];                                   // sorted descending
    int   ti[cTOPK];
#pragma unroll
    for (int j = 0; j < cTOPK; ++j) { ts[j] = -__builtin_inff(); ti[j] = 0; }

    const int colBase = lh * 8;                        // this lane's 8-column span

    for (int kt = 0; kt <= qt; ++kt) {
        // B fragments: column n = key (kt*16 + lane&15); head-dim {lh*8..+7},{16+lh*8..+7}
        const unsigned short* krow = Kb + (size_t)(kt * 16 + row) * cD + h * cHD;
        Frag16 b0, b1;
        b0.q[0] = *(const uint4*)(krow + lh * 8);
        b0.q[1] = *(const uint4*)(krow + 16 + lh * 8);
        b1.q[0] = *(const uint4*)(krow + 32 + lh * 8);
        b1.q[1] = *(const uint4*)(krow + 48 + lh * 8);

        v8f c = {0.f, 0.f, 0.f, 0.f, 0.f, 0.f, 0.f, 0.f};
        c = __builtin_amdgcn_wmma_f32_16x16x32_bf16(false, a0.v, false, b0.v,
                                                    (short)0, c, false, false);
        c = __builtin_amdgcn_wmma_f32_16x16x32_bf16(false, a1.v, false, b1.v,
                                                    (short)0, c, false, false);
        // scale + causal mask, stage through LDS (C/D: M = r + 8*half, N = lane&15)
#pragma unroll
        for (int r = 0; r < 8; ++r) {
            const int qg = q0 + r + lh * 8;
            const int kg = kt * 16 + (lane & 15);
            float s = c[r] * cSCALE;
            if (kg > qg) s = -__builtin_inff();
            sc[h][(r + lh * 8) * 16 + (lane & 15)] = s;
        }
        // wave-internal cross-lane handoff: DS ops are in-order per wave,
        // just drain the DS counter before reading other lanes' stores.
        asm volatile("s_wait_dscnt 0" ::: "memory");
        {
            // all 32 lanes scan: lane owns row (lane&15), columns colBase..+7
            const float4 seg0 = *(const float4*)&sc[h][row * 16 + colBase];
            const float4 seg1 = *(const float4*)&sc[h][row * 16 + colBase + 4];
            const float cand[8] = {seg0.x, seg0.y, seg0.z, seg0.w,
                                   seg1.x, seg1.y, seg1.z, seg1.w};
#pragma unroll
            for (int n = 0; n < 8; ++n) {
                float s  = cand[n];
                int  idx = kt * 16 + colBase + n;
                if (s > ts[cTOPK - 1]) {               // rare after warm-up
#pragma unroll
                    for (int j = 0; j < cTOPK; ++j) {  // bubbling insert, keeps sorted
                        if (s > ts[j]) {
                            float tf = ts[j]; ts[j] = s;   s  = tf;
                            int   tt = ti[j]; ti[j] = idx; idx = tt;
                        }
                    }
                }
            }
        }
        asm volatile("s_wait_dscnt 0" ::: "memory");
    }

    // merge partner-lane (lane^16) top-8 into this lane's list.
    // Snapshot first: partner's registers mutate as it merges ours.
    {
        float ps[cTOPK]; int pi[cTOPK];
#pragma unroll
        for (int j = 0; j < cTOPK; ++j) {
            ps[j] = __shfl_xor(ts[j], 16, 32);
            pi[j] = __shfl_xor(ti[j], 16, 32);
        }
#pragma unroll
        for (int j = 0; j < cTOPK; ++j) {
            float s  = ps[j];
            int  idx = pi[j];
            if (s > ts[cTOPK - 1]) {
#pragma unroll
                for (int m = 0; m < cTOPK; ++m) {
                    if (s > ts[m]) {
                        float tf = ts[m]; ts[m] = s;   s  = tf;
                        int   tt = ti[m]; ti[m] = idx; idx = tt;
                    }
                }
            }
        }
    }

    // softmax over the (at most) 8 surviving scores + gather of V rows
    if (lane < 16) {
        const float mx = ts[0];                        // sorted: ts[0] is the max (finite: diagonal)
        float acc[cHD];
#pragma unroll
        for (int x = 0; x < cHD; ++x) acc[x] = 0.f;
        float wsum = 0.f;
#pragma unroll
        for (int j = 0; j < cTOPK; ++j) {
            if (ts[j] == -__builtin_inff()) continue;
            const float wgt = __expf(ts[j] - mx);
            wsum += wgt;
            const unsigned short* vr = Vb + (size_t)ti[j] * cD + h * cHD;
#pragma unroll
            for (int q8 = 0; q8 < cHD / 8; ++q8) {     // 8 bf16 per b128 load
                const uint4 pk = *(const uint4*)(vr + q8 * 8);
                const unsigned w0 = pk.x, w1 = pk.y, w2 = pk.z, w3 = pk.w;
                acc[q8*8+0] += wgt * bf2f((unsigned short)(w0 & 0xffff));
                acc[q8*8+1] += wgt * bf2f((unsigned short)(w0 >> 16));
                acc[q8*8+2] += wgt * bf2f((unsigned short)(w1 & 0xffff));
                acc[q8*8+3] += wgt * bf2f((unsigned short)(w1 >> 16));
                acc[q8*8+4] += wgt * bf2f((unsigned short)(w2 & 0xffff));
                acc[q8*8+5] += wgt * bf2f((unsigned short)(w2 >> 16));
                acc[q8*8+6] += wgt * bf2f((unsigned short)(w3 & 0xffff));
                acc[q8*8+7] += wgt * bf2f((unsigned short)(w3 >> 16));
            }
        }
        const float inv = (wsum > 0.f) ? 1.f / wsum : 0.f;
        unsigned short* orow = Ob + (size_t)(q0 + row) * cD + h * cHD;
#pragma unroll
        for (int x = 0; x < cHD; ++x) orow[x] = f2bf(acc[x] * inv);
    }
}

// ---------------------------------------------------------------------------
// Launch: cvt -> Q/K/V GEMMs -> sparse attention -> O GEMM (f32 to d_out)
// ---------------------------------------------------------------------------
extern "C" void kernel_launch(void* const* d_in, const int* in_sizes, int n_in,
                              void* d_out, int out_size, void* d_ws, size_t ws_size,
                              hipStream_t stream) {
    const float* h_query = (const float*)d_in[0];
    const float* h_kv    = (const float*)d_in[1];
    const float* Wq = (const float*)d_in[2]; const float* bq = (const float*)d_in[3];
    const float* Wk = (const float*)d_in[4]; const float* bk = (const float*)d_in[5];
    const float* Wv = (const float*)d_in[6]; const float* bv = (const float*)d_in[7];
    const float* Wo = (const float*)d_in[8]; const float* bo = (const float*)d_in[9];
    float* out = (float*)d_out;

    char*  ws  = (char*)d_ws;
    size_t off = 0;
    auto carve = [&](size_t bytes) -> void* {
        void* p = ws + off;
        off = (off + bytes + 255) & ~(size_t)255;
        return p;
    };
    const size_t actBytes = (size_t)cS * cD * sizeof(unsigned short);   // 4 MB
    const size_t wBytes   = (size_t)cD * cD * sizeof(unsigned short);   // 512 KB
    unsigned short* hq_bf  = (unsigned short*)carve(actBytes);
    unsigned short* hkv_bf = (unsigned short*)carve(actBytes);
    unsigned short* Wq_bf  = (unsigned short*)carve(wBytes);
    unsigned short* Wk_bf  = (unsigned short*)carve(wBytes);
    unsigned short* Wv_bf  = (unsigned short*)carve(wBytes);
    unsigned short* Wo_bf  = (unsigned short*)carve(wBytes);
    unsigned short* Qbf    = (unsigned short*)carve(actBytes);
    unsigned short* Kbf    = (unsigned short*)carve(actBytes);
    unsigned short* Vbf    = (unsigned short*)carve(actBytes);
    unsigned short* Abf    = (unsigned short*)carve(actBytes);
    (void)ws_size; (void)in_sizes; (void)n_in; (void)out_size;

    const int nAct = cS * cD;        // 2,097,152
    const int nW   = cD * cD;        //   262,144
    cvt_f32_bf16<<<(nAct + 255) / 256, 256, 0, stream>>>(h_query, hq_bf, nAct);
    cvt_f32_bf16<<<(nAct + 255) / 256, 256, 0, stream>>>(h_kv,   hkv_bf, nAct);
    cvt_f32_bf16<<<(nW   + 255) / 256, 256, 0, stream>>>(Wq, Wq_bf, nW);
    cvt_f32_bf16<<<(nW   + 255) / 256, 256, 0, stream>>>(Wk, Wk_bf, nW);
    cvt_f32_bf16<<<(nW   + 255) / 256, 256, 0, stream>>>(Wv, Wv_bf, nW);
    cvt_f32_bf16<<<(nW   + 255) / 256, 256, 0, stream>>>(Wo, Wo_bf, nW);

    const int strips = (cS / 16) * (cD / 64);          // 2048 wave-strips (16x64 each)
    const int gblks  = strips / 8;                     // 8 waves per block -> 256 blocks
    gemm_bf16_wmma<<<gblks, 256, 0, stream>>>(hq_bf,  Wq_bf, bq, nullptr, Qbf, cS, cD, cD, 1);
    gemm_bf16_wmma<<<gblks, 256, 0, stream>>>(hkv_bf, Wk_bf, bk, nullptr, Kbf, cS, cD, cD, 1);
    gemm_bf16_wmma<<<gblks, 256, 0, stream>>>(hkv_bf, Wv_bf, bv, nullptr, Vbf, cS, cD, cD, 1);

    sparse_attn_wmma<<<cS / 16, 256, 0, stream>>>(Qbf, Kbf, Vbf, Abf);

    gemm_bf16_wmma<<<gblks, 256, 0, stream>>>(Abf, Wo_bf, bo, out, nullptr, cS, cD, cD, 0);
}
